// quant_dense_45784351375737
// MI455X (gfx1250) — compile-verified, matrix operand order checked by reading, not run on hardware
//
#include <hip/hip_runtime.h>
#include <stdint.h>

typedef __bf16 bf16_t;
typedef __attribute__((ext_vector_type(16))) __bf16 v16bf;
typedef __attribute__((ext_vector_type(8)))  __bf16 v8bf;
typedef __attribute__((ext_vector_type(4)))  __bf16 v4bf;
typedef __attribute__((ext_vector_type(8)))  float  v8f;

union FragBF16 {
    v16bf v;
    v8bf  h[2];
};

// LDS tile geometry: 128 rows x 32 k, row stride padded to 40 elements (80 B,
// 16B-aligned, conflict-free half-wave ds_load_b128 pattern).
#define LDS_ROW  40
#define TILE_ELE (128 * LDS_ROW)

// ---------------------------------------------------------------------------
// CDNA5 async global->LDS copy (ASYNCcnt path, no VGPR staging).
// GVS addressing: mem = SGPR64 base + VGPR32 byte offset.
// ---------------------------------------------------------------------------
__device__ __forceinline__ void async_ld16(const void* sbase, uint32_t goff_bytes,
                                           uint32_t lds_byte_addr) {
    asm volatile("global_load_async_to_lds_b128 %0, %1, %2 offset:0"
                 :: "v"(lds_byte_addr), "v"(goff_bytes), "s"(sbase)
                 : "memory");
}
__device__ __forceinline__ void wait_async0() {
    asm volatile("s_wait_asynccnt 0" ::: "memory");
}

// ---------------------------------------------------------------------------
// Phase 1a: split x (f32) into x_hi + x_lo bf16 (exact-sum 2-term split).
// ---------------------------------------------------------------------------
__global__ __launch_bounds__(256)
void convert_x_kernel(const float* __restrict__ x,
                      bf16_t* __restrict__ xhi,
                      bf16_t* __restrict__ xlo,
                      long long n4) {
    long long i = (long long)blockIdx.x * blockDim.x + threadIdx.x;
    if (i >= n4) return;
    const float4 xv = ((const float4*)x)[i];
    float f[4] = {xv.x, xv.y, xv.z, xv.w};
    v4bf hi, lo;
#pragma unroll
    for (int j = 0; j < 4; ++j) {
        __bf16 h = (__bf16)f[j];
        hi[j] = h;
        lo[j] = (__bf16)(f[j] - (float)h);
    }
    *(v4bf*)(xhi + 4 * i) = hi;
    *(v4bf*)(xlo + 4 * i) = lo;
}

// ---------------------------------------------------------------------------
// Phase 1b: sT[n][k] = sign(W[k][n]) as bf16 (exact), LDS-tiled transpose.
// ---------------------------------------------------------------------------
__global__ __launch_bounds__(256)
void sign_transpose_kernel(const float* __restrict__ W,
                           bf16_t* __restrict__ sT,
                           int K, int N) {
    __shared__ bf16_t tile[32][33];
    const int n0 = blockIdx.x * 32;
    const int k0 = blockIdx.y * 32;
    const int tx = threadIdx.x;   // 0..31
    const int ty = threadIdx.y;   // 0..7
#pragma unroll
    for (int j = 0; j < 4; ++j) {
        int k = k0 + ty + 8 * j;
        float w = W[(size_t)k * N + (n0 + tx)];
        float s = (w > 0.f) ? 1.f : ((w < 0.f) ? -1.f : 0.f);
        tile[ty + 8 * j][tx] = (bf16_t)s;
    }
    __syncthreads();
#pragma unroll
    for (int j = 0; j < 4; ++j) {
        int n = n0 + ty + 8 * j;
        sT[(size_t)n * K + (k0 + tx)] = tile[tx][ty + 8 * j];
    }
}

// ---------------------------------------------------------------------------
// Phase 2: y = (x_hi + x_lo) @ sT^T via V_WMMA_F32_16X16X32_BF16.
// 256 threads = 8 wave32; block tile 128x128; wave tile 32x64 (2x4 subtiles).
// Double-buffered LDS staging fed by GLOBAL_LOAD_ASYNC_TO_LDS_B128.
// ---------------------------------------------------------------------------
__global__ __launch_bounds__(256)
void gemm_sign_kernel(const bf16_t* __restrict__ xhi,
                      const bf16_t* __restrict__ xlo,
                      const bf16_t* __restrict__ sT,
                      float* __restrict__ out,
                      int M, int N, int K) {
    // [buffer][matrix: Ahi,Alo,B][tile]
    __shared__ bf16_t smem[2][3][TILE_ELE];

    const int tid  = threadIdx.x;
    const int wave = tid >> 5;
    const int lane = tid & 31;
    const int half = lane >> 4;
    const int l16  = lane & 15;
    const int wr   = wave >> 1;   // 0..3 : 32-row group
    const int wc   = wave & 1;    // 0..1 : 64-col group

    const int m_blk = blockIdx.y * 128;
    const int n_blk = blockIdx.x * 128;

    // Low 32 bits of a generic pointer into LDS == wave-relative LDS address.
    const uint32_t lds0 = (uint32_t)(uintptr_t)&smem[0][0][0];

    // 512 16B-chunks per 128x32 tile -> 2 chunks per thread per matrix.
    auto issue = [&](int k0, int buf) {
        const uint32_t lb = lds0 + (uint32_t)buf * (3u * TILE_ELE * 2u);
#pragma unroll
        for (int j = 0; j < 2; ++j) {
            const int chunk = tid + j * 256;
            const int r = chunk >> 2;        // tile row / col
            const int c = chunk & 3;         // 8-element k chunk
            const uint32_t lo = lb + (uint32_t)(r * LDS_ROW + c * 8) * 2u;
            const uint32_t ga = (uint32_t)(((m_blk + r) * K + k0 + c * 8) * 2);
            const uint32_t gb = (uint32_t)(((n_blk + r) * K + k0 + c * 8) * 2);
            async_ld16(xhi, ga, lo);
            async_ld16(xlo, ga, lo + (uint32_t)(TILE_ELE * 2));
            async_ld16(sT,  gb, lo + (uint32_t)(2 * TILE_ELE * 2));
        }
    };

    v8f acc[2][4] = {};

    issue(0, 0);
    wait_async0();
    __syncthreads();

    const int T = K / 32;
    for (int t = 0; t < T; ++t) {
        const int buf = t & 1;
        if (t + 1 < T) issue((t + 1) * 32, buf ^ 1);

        const bf16_t* sAh = &smem[buf][0][0];
        const bf16_t* sAl = &smem[buf][1][0];
        const bf16_t* sB  = &smem[buf][2][0];

        // Fragment loads (ISA 16-bit A layout: lane<16 -> K {0..7,16..23},
        // lane>=16 -> K {8..15,24..31}; row/col = lane&15).
        FragBF16 ahi[2], alo[2], b[4];
#pragma unroll
        for (int im = 0; im < 2; ++im) {
            const int off = (wr * 32 + im * 16 + l16) * LDS_ROW + half * 8;
            ahi[im].h[0] = *(const v8bf*)&sAh[off];
            ahi[im].h[1] = *(const v8bf*)&sAh[off + 16];
            alo[im].h[0] = *(const v8bf*)&sAl[off];
            alo[im].h[1] = *(const v8bf*)&sAl[off + 16];
        }
#pragma unroll
        for (int jn = 0; jn < 4; ++jn) {
            const int off = (wc * 64 + jn * 16 + l16) * LDS_ROW + half * 8;
            b[jn].h[0] = *(const v8bf*)&sB[off];
            b[jn].h[1] = *(const v8bf*)&sB[off + 16];
        }

#pragma unroll
        for (int im = 0; im < 2; ++im) {
#pragma unroll
            for (int jn = 0; jn < 4; ++jn) {
                acc[im][jn] = __builtin_amdgcn_wmma_f32_16x16x32_bf16(
                    false, ahi[im].v, false, b[jn].v,
                    (short)0, acc[im][jn], false, false);
                acc[im][jn] = __builtin_amdgcn_wmma_f32_16x16x32_bf16(
                    false, alo[im].v, false, b[jn].v,
                    (short)0, acc[im][jn], false, false);
            }
        }

        wait_async0();     // next tile fully landed in LDS
        __syncthreads();   // all waves done reading current buffer
    }

    // C/D layout: VGPR v, lane<16 -> M = v, lane>=16 -> M = 8+v; N = lane&15.
#pragma unroll
    for (int im = 0; im < 2; ++im) {
        const int rbase = m_blk + wr * 32 + im * 16 + (half << 3);
#pragma unroll
        for (int jn = 0; jn < 4; ++jn) {
            const int col = n_blk + wc * 64 + jn * 16 + l16;
#pragma unroll
            for (int v = 0; v < 8; ++v) {
                out[(size_t)(rbase + v) * N + col] = acc[im][jn][v];
            }
        }
    }
}

// ---------------------------------------------------------------------------
extern "C" void kernel_launch(void* const* d_in, const int* in_sizes, int n_in,
                              void* d_out, int out_size, void* d_ws, size_t ws_size,
                              hipStream_t stream) {
    const float* x = (const float*)d_in[0];
    const float* W = (const float*)d_in[1];
    float* out = (float*)d_out;

    const int K = 4096;
    const int N = 4096;
    const int M = in_sizes[0] / K;   // 8192

    // Workspace layout (167.5 MB total; fits global L2 of 192 MB):
    //   sT  : N*K bf16  (33.5 MB)   sT[n][k] = sign(W[k][n])
    //   xhi : M*K bf16  (67 MB)
    //   xlo : M*K bf16  (67 MB)
    char* ws = (char*)d_ws;
    bf16_t* sTp = (bf16_t*)ws;
    bf16_t* xhi = (bf16_t*)(ws + (size_t)N * K * sizeof(bf16_t));
    bf16_t* xlo = xhi + (size_t)M * K;

    long long n4 = (long long)M * K / 4;
    convert_x_kernel<<<dim3((unsigned)((n4 + 255) / 256)), 256, 0, stream>>>(x, xhi, xlo, n4);

    dim3 tgrid(N / 32, K / 32);
    dim3 tblk(32, 8);
    sign_transpose_kernel<<<tgrid, tblk, 0, stream>>>(W, sTp, K, N);

    dim3 ggrid(N / 128, M / 128);
    gemm_sign_kernel<<<ggrid, 256, 0, stream>>>(xhi, xlo, sTp, out, M, N, K);
}
